// EmbeddingAlignerOT_34617436406388
// MI455X (gfx1250) — compile-verified, matrix operand order checked by reading, not run on hardware
//
#include <hip/hip_runtime.h>
#include <math.h>

// ---------------------------------------------------------------------------
// EmbeddingAlignerOT for MI455X (gfx1250, wave32, WMMA)
// Shapes fixed by setup_inputs(): B=32, S=512, G=512, SH=GH=1024
// GEMMs: bf16 WMMA, f32 accum, 64x32 register tile per wave (4x2 subtiles,
//        8 wmma per 6 fragment loads -> ~21 flop/byte from L1/L2).
// Sinkhorn: K (33.5MB) + K^T (33.5MB) both resident in the 192MB L2.
// ---------------------------------------------------------------------------

typedef __bf16 bf16_t;
typedef __attribute__((ext_vector_type(16))) __bf16 v16bf;
typedef __attribute__((ext_vector_type(8)))  __bf16 v8bf;
typedef __attribute__((ext_vector_type(8)))  float  v8f;

constexpr int B_ = 32;
constexpr int S_ = 512;
constexpr int G_ = 512;
constexpr int D_ = 1024;   // SH
constexpr int H_ = 1024;   // GH

constexpr float OT_REG   = 0.1f;
constexpr float OT_REG_M = 10.0f;
constexpr int   N_ITER   = 100;
constexpr float FI       = OT_REG_M / (OT_REG_M + OT_REG); // 0.990099f
constexpr float EPSF     = 1e-12f;

// ----------------------------- WMMA helpers --------------------------------
// 16-bit A (16x32) / B (32x16) fragment per ISA 7.12.2: per lane, elements
// 0..7 are K-contiguous at k0 + 8*(lane>=16), elements 8..15 at +16 more;
// row/col = lane&15. Both chunks 16B-aligned -> two b128 loads per fragment.
__device__ __forceinline__ v16bf load_frag(const bf16_t* tile, int stride, int lane) {
  const bf16_t* p = tile + (size_t)(lane & 15) * stride + ((lane >> 4) << 3);
  v8bf lo = *reinterpret_cast<const v8bf*>(p);
  v8bf hi = *reinterpret_cast<const v8bf*>(p + 16);
  v16bf f;
#pragma unroll
  for (int i = 0; i < 8; ++i) { f[i] = lo[i]; f[i + 8] = hi[i]; }
  return f;
}

// 64x32 per-wave register-tiled bf16 GEMM core: acc[i][j] += A(64xK) * B^T(Kx32)
// Abase points at row m0, Bbase at row n0 (both K-major, leading dim lda/ldb).
__device__ __forceinline__ void gemm_64x32(const bf16_t* __restrict__ Abase, int lda,
                                           const bf16_t* __restrict__ Bbase, int ldb,
                                           int K, int lane, v8f acc[4][2]) {
  for (int k = 0; k < K; k += 32) {
    v16bf af[4], bf[2];
#pragma unroll
    for (int i = 0; i < 4; ++i) af[i] = load_frag(Abase + (size_t)(16 * i) * lda + k, lda, lane);
#pragma unroll
    for (int j = 0; j < 2; ++j) bf[j] = load_frag(Bbase + (size_t)(16 * j) * ldb + k, ldb, lane);
#pragma unroll
    for (int i = 0; i < 4; ++i)
#pragma unroll
      for (int j = 0; j < 2; ++j)
        acc[i][j] = __builtin_amdgcn_wmma_f32_16x16x32_bf16(false, af[i], false, bf[j],
                                                            (short)0, acc[i][j], false, false);
  }
}

// C/D 16x16 f32 layout: register r -> row = r + 8*(lane>=16), col = lane & 15.

// ------------------------- elementwise conversion --------------------------
__global__ void cvt_bf16_kernel(const float* __restrict__ x, bf16_t* __restrict__ y, int n) {
  int i = blockIdx.x * blockDim.x + threadIdx.x;
  int stride = gridDim.x * blockDim.x;
  for (; i < n; i += stride) y[i] = (bf16_t)x[i];
}

// g [B,G,H] f32 -> gT [B,H,G] bf16 (tiled, coalesced both sides)
__global__ void transpose_g_kernel(const float* __restrict__ g, bf16_t* __restrict__ gT) {
  __shared__ float tile[16][17];
  int b  = blockIdx.z;
  int g0 = blockIdx.x * 16;
  int h0 = blockIdx.y * 16;
  int tx = threadIdx.x, ty = threadIdx.y;
  tile[ty][tx] = g[((size_t)b * G_ + g0 + ty) * H_ + h0 + tx];
  __syncthreads();
  gT[((size_t)b * H_ + h0 + ty) * G_ + g0 + tx] = (bf16_t)tile[tx][ty];
}

// --------------------------- linear: t = src@W^T + b -----------------------
// grid(B*S/128, H/128), block 256 (8 waves): 128x128 macro-tile, 64x32/wave.
__global__ void linear_wmma_kernel(const bf16_t* __restrict__ A, const bf16_t* __restrict__ Wb,
                                   const float* __restrict__ bias, bf16_t* __restrict__ tout) {
  int lane = threadIdx.x & 31;
  int wave = threadIdx.x >> 5;
  int m0 = blockIdx.x * 128 + (wave & 1) * 64;
  int n0 = blockIdx.y * 128 + (wave >> 1) * 32;
  v8f acc[4][2] = {};
  gemm_64x32(A + (size_t)m0 * D_, D_, Wb + (size_t)n0 * D_, D_, D_, lane, acc);
  int col = lane & 15, rbase = (lane >> 4) << 3;
#pragma unroll
  for (int j = 0; j < 2; ++j) {
    float bv = bias[n0 + 16 * j + col];
#pragma unroll
    for (int i = 0; i < 4; ++i)
#pragma unroll
      for (int r = 0; r < 8; ++r)
        tout[((size_t)(m0 + 16 * i + rbase + r)) * H_ + n0 + 16 * j + col] =
            (bf16_t)(acc[i][j][r] + bv);
  }
}

// ------------------------------ row norms ----------------------------------
__global__ void rownorm_bf16_kernel(const bf16_t* __restrict__ x, const int* __restrict__ mask,
                                    float* __restrict__ s2o, float* __restrict__ wraw,
                                    int nrows, int ncols) {
  int row  = blockIdx.x * (blockDim.x >> 5) + (threadIdx.x >> 5);
  int lane = threadIdx.x & 31;
  if (row >= nrows) return;
  const bf16_t* p = x + (size_t)row * ncols;
  float acc = 0.f;
  for (int j = lane; j < ncols; j += 32) { float v = (float)p[j]; acc = fmaf(v, v, acc); }
#pragma unroll
  for (int off = 16; off > 0; off >>= 1) acc += __shfl_xor(acc, off, 32);
  if (lane == 0) {
    s2o[row] = acc;
    float mags = sqrtf(acc) + EPSF;
    wraw[row] = (float)mask[row] * mags + EPSF;
  }
}

__global__ void rownorm_f32_kernel(const float* __restrict__ x, const int* __restrict__ mask,
                                   float* __restrict__ s2o, float* __restrict__ wraw,
                                   int nrows, int ncols) {
  int row  = blockIdx.x * (blockDim.x >> 5) + (threadIdx.x >> 5);
  int lane = threadIdx.x & 31;
  if (row >= nrows) return;
  const float* p = x + (size_t)row * ncols;
  float acc = 0.f;
  for (int j = lane; j < ncols; j += 32) { float v = p[j]; acc = fmaf(v, v, acc); }
#pragma unroll
  for (int off = 16; off > 0; off >>= 1) acc += __shfl_xor(acc, off, 32);
  if (lane == 0) {
    s2o[row] = acc;
    float mags = sqrtf(acc) + EPSF;
    wraw[row] = (float)mask[row] * mags + EPSF;
  }
}

// per-batch normalization: w = wraw / max(sum(wraw), 1e-12); grid=B, block=512
__global__ void normalize_w_kernel(const float* __restrict__ wraw, float* __restrict__ wout) {
  __shared__ float red[512];
  int b = blockIdx.x, tid = threadIdx.x;
  float v = wraw[b * 512 + tid];
  red[tid] = v;
  __syncthreads();
#pragma unroll
  for (int off = 256; off > 0; off >>= 1) {
    if (tid < off) red[tid] += red[tid + off];
    __syncthreads();
  }
  wout[b * 512 + tid] = v / fmaxf(red[0], 1e-12f);
}

// ------------------- cost + Gibbs kernel: K, K^T ---------------------------
// dot = t.g via WMMA; cost = sqrt(max(s2+g2-2dot,0))*sw*gw + eps; K=exp(-cost/reg)
// grid(S/128, G/128, B), block 256
__global__ void cost_wmma_kernel(const bf16_t* __restrict__ t, const bf16_t* __restrict__ g,
                                 const float* __restrict__ s2, const float* __restrict__ g2,
                                 const float* __restrict__ sw, const float* __restrict__ gw,
                                 float* __restrict__ Kmat, float* __restrict__ KT) {
  int lane = threadIdx.x & 31;
  int wave = threadIdx.x >> 5;
  int b  = blockIdx.z;
  int s0 = blockIdx.x * 128 + (wave & 1) * 64;
  int g0 = blockIdx.y * 128 + (wave >> 1) * 32;
  v8f acc[4][2] = {};
  gemm_64x32(t + ((size_t)b * S_ + s0) * H_, H_,
             g + ((size_t)b * G_ + g0) * H_, H_, H_, lane, acc);
  int col = lane & 15, rbase = (lane >> 4) << 3;
#pragma unroll
  for (int j = 0; j < 2; ++j) {
    int gi = g0 + 16 * j + col;
    float g2v = g2[b * G_ + gi];
    float gwv = gw[b * G_ + gi];
#pragma unroll
    for (int i = 0; i < 4; ++i)
#pragma unroll
      for (int r = 0; r < 8; ++r) {
        int si = s0 + 16 * i + rbase + r;
        float sq = s2[b * S_ + si] + g2v - 2.f * acc[i][j][r];
        float d  = sqrtf(fmaxf(sq, 0.f));
        float c  = d * sw[b * S_ + si] * gwv + EPSF;
        float kv = __expf(-c * (1.0f / OT_REG));
        Kmat[((size_t)b * S_ + si) * G_ + gi] = kv;
        KT  [((size_t)b * G_ + gi) * S_ + si] = kv;
      }
  }
}

// ------------------------------- Sinkhorn ----------------------------------
// one 512-thread WG per batch; K & K^T stay hot in L2; u/v/a/b in LDS.
__global__ void sinkhorn_kernel(const float* __restrict__ Kmat, const float* __restrict__ KT,
                                const float* __restrict__ sw, const float* __restrict__ gw,
                                float* __restrict__ u_g, float* __restrict__ v_g) {
  __shared__ float u_l[S_], v_l[G_], a_l[S_], b_l[G_];
  int b = blockIdx.x, tid = threadIdx.x;
  a_l[tid] = sw[b * S_ + tid];
  b_l[tid] = gw[b * G_ + tid];
  u_l[tid] = 1.0f / S_;
  v_l[tid] = 1.0f / G_;
  __syncthreads();
  const float* Kb  = Kmat + (size_t)b * S_ * G_;
  const float* KTb = KT   + (size_t)b * S_ * G_;
  for (int it = 0; it < N_ITER; ++it) {
    // Kv[s] = sum_g K[s,g] v[g]  (read K^T: coalesced over tid=s)
    float acc = 0.f;
#pragma unroll 8
    for (int gg = 0; gg < G_; ++gg) acc = fmaf(KTb[(size_t)gg * S_ + tid], v_l[gg], acc);
    float un = __powf(a_l[tid] / acc, FI);
    __syncthreads();
    u_l[tid] = un;
    __syncthreads();
    // Ktu[g] = sum_s K[s,g] u[s]  (read K: coalesced over tid=g)
    float acc2 = 0.f;
#pragma unroll 8
    for (int ss = 0; ss < S_; ++ss) acc2 = fmaf(Kb[(size_t)ss * G_ + tid], u_l[ss], acc2);
    float vn = __powf(b_l[tid] / acc2, FI);
    __syncthreads();
    v_l[tid] = vn;
    __syncthreads();
  }
  u_g[b * S_ + tid] = u_l[tid];
  v_g[b * G_ + tid] = v_l[tid];
}

// P[b,s,g] = bf16(u[b,s] * K[b,s,g] * v[b,g])
__global__ void plan_scale_kernel(const float* __restrict__ Kmat, const float* __restrict__ u_g,
                                  const float* __restrict__ v_g, bf16_t* __restrict__ P) {
  size_t n = (size_t)B_ * S_ * G_;
  size_t i = (size_t)blockIdx.x * blockDim.x + threadIdx.x;
  size_t stride = (size_t)gridDim.x * blockDim.x;
  for (; i < n; i += stride) {
    int gg = (int)(i % G_);
    size_t sg = i / G_;
    int ss = (int)(sg % S_);
    int b  = (int)(sg / S_);
    P[i] = (bf16_t)(u_g[b * S_ + ss] * Kmat[i] * v_g[b * G_ + gg]);
  }
}

// aligned[b,s,h] = sum_g P[b,s,g] * gT[b,h,g]   (both K-contiguous, bf16 WMMA)
// grid(S/128, H/128, B), block 256
__global__ void aligned_wmma_kernel(const bf16_t* __restrict__ P, const bf16_t* __restrict__ gT,
                                    float* __restrict__ out) {
  int lane = threadIdx.x & 31;
  int wave = threadIdx.x >> 5;
  int b  = blockIdx.z;
  int s0 = blockIdx.x * 128 + (wave & 1) * 64;
  int h0 = blockIdx.y * 128 + (wave >> 1) * 32;
  v8f acc[4][2] = {};
  gemm_64x32(P + ((size_t)b * S_ + s0) * G_, G_,
             gT + ((size_t)b * H_ + h0) * G_, G_, G_, lane, acc);
  int col = lane & 15, rbase = (lane >> 4) << 3;
#pragma unroll
  for (int j = 0; j < 2; ++j)
#pragma unroll
    for (int i = 0; i < 4; ++i)
#pragma unroll
      for (int r = 0; r < 8; ++r)
        out[((size_t)b * S_ + s0 + 16 * i + rbase + r) * H_ + h0 + 16 * j + col] = acc[i][j][r];
}

// ------------------------------- launcher ----------------------------------
extern "C" void kernel_launch(void* const* d_in, const int* in_sizes, int n_in,
                              void* d_out, int out_size, void* d_ws, size_t ws_size,
                              hipStream_t stream) {
  const float* src   = (const float*)d_in[0];  // [B,S,D]
  const float* tgt   = (const float*)d_in[1];  // [B,G,H]
  const int*   smask = (const int*)d_in[2];    // [B,S]
  const int*   gmask = (const int*)d_in[3];    // [B,G]
  const float* W     = (const float*)d_in[4];  // [H,D]
  const float* bias  = (const float*)d_in[5];  // [H]
  float* out = (float*)d_out;                  // [B,S,H]

  // workspace layout (256B aligned)
  char* ws = (char*)d_ws;
  size_t o = 0;
  auto take = [&](size_t bytes) { char* p = ws + o; o = (o + bytes + 255) & ~(size_t)255; return p; };
  bf16_t* src_bf = (bf16_t*)take((size_t)B_ * S_ * D_ * 2);
  bf16_t* W_bf   = (bf16_t*)take((size_t)H_ * D_ * 2);
  bf16_t* g_bf   = (bf16_t*)take((size_t)B_ * G_ * H_ * 2);
  bf16_t* gT_bf  = (bf16_t*)take((size_t)B_ * H_ * G_ * 2);
  bf16_t* t_bf   = (bf16_t*)take((size_t)B_ * S_ * H_ * 2);
  float*  Kmat   = (float*) take((size_t)B_ * S_ * G_ * 4);
  float*  KT     = (float*) take((size_t)B_ * S_ * G_ * 4);
  bf16_t* P_bf   = (bf16_t*)take((size_t)B_ * S_ * G_ * 2);
  float*  s2     = (float*) take((size_t)B_ * S_ * 4);
  float*  g2     = (float*) take((size_t)B_ * G_ * 4);
  float*  wraw_s = (float*) take((size_t)B_ * S_ * 4);
  float*  wraw_g = (float*) take((size_t)B_ * G_ * 4);
  float*  sw     = (float*) take((size_t)B_ * S_ * 4);
  float*  gw     = (float*) take((size_t)B_ * G_ * 4);
  float*  u_g    = (float*) take((size_t)B_ * S_ * 4);
  float*  v_g    = (float*) take((size_t)B_ * G_ * 4);
  (void)ws_size; (void)in_sizes; (void)n_in; (void)out_size;

  // 1) f32 -> bf16 staging
  cvt_bf16_kernel<<<4096, 256, 0, stream>>>(src, src_bf, B_ * S_ * D_);
  cvt_bf16_kernel<<<1024, 256, 0, stream>>>(W, W_bf, H_ * D_);
  cvt_bf16_kernel<<<4096, 256, 0, stream>>>(tgt, g_bf, B_ * G_ * H_);
  transpose_g_kernel<<<dim3(G_ / 16, H_ / 16, B_), dim3(16, 16), 0, stream>>>(tgt, gT_bf);

  // 2) t = src @ W^T + b  (bf16 WMMA, f32 accum, register-tiled)
  linear_wmma_kernel<<<dim3(B_ * S_ / 128, H_ / 128), 256, 0, stream>>>(src_bf, W_bf, bias, t_bf);

  // 3) norms + weights
  rownorm_bf16_kernel<<<B_ * S_ / 8, 256, 0, stream>>>(t_bf, smask, s2, wraw_s, B_ * S_, H_);
  rownorm_f32_kernel<<<B_ * G_ / 8, 256, 0, stream>>>(tgt, gmask, g2, wraw_g, B_ * G_, H_);
  normalize_w_kernel<<<B_, 512, 0, stream>>>(wraw_s, sw);
  normalize_w_kernel<<<B_, 512, 0, stream>>>(wraw_g, gw);

  // 4) cost + Gibbs kernel K and K^T (both resident in 192MB L2 afterwards)
  cost_wmma_kernel<<<dim3(S_ / 128, G_ / 128, B_), 256, 0, stream>>>(t_bf, g_bf, s2, g2, sw, gw,
                                                                     Kmat, KT);

  // 5) 100 unbalanced Sinkhorn iterations, L2-resident
  sinkhorn_kernel<<<B_, 512, 0, stream>>>(Kmat, KT, sw, gw, u_g, v_g);

  // 6) plan = diag(u) K diag(v) -> bf16, then aligned = plan @ g (bf16 WMMA)
  plan_scale_kernel<<<8192, 256, 0, stream>>>(Kmat, u_g, v_g, P_bf);
  aligned_wmma_kernel<<<dim3(S_ / 128, H_ / 128, B_), 256, 0, stream>>>(P_bf, gT_bf, out);
}
